// TransitionDown_62199716381216
// MI455X (gfx1250) — matrix operand order repaired and checked by reading, not compiled, not run on hardware
//
#include <hip/hip_runtime.h>

// ---------------- problem constants (from reference setup_inputs) ------------
#define BB 4
#define NN 4096
#define FF 128
#define SS 1024
#define KK 64
#define CIN 256
#define COUT 256
#define ROWS ((long)BB*SS*KK)     // 262144
#define EPSBN 1e-5f

// ---------------- types for WMMA / TDM --------------------------------------
typedef __attribute__((ext_vector_type(16))) __bf16 v16bf;
typedef __attribute__((ext_vector_type(8)))  float  v8f;
typedef __attribute__((ext_vector_type(4)))  unsigned u32x4;
typedef __attribute__((ext_vector_type(8)))  int      i32x8;
typedef __attribute__((ext_vector_type(4)))  int      i32x4;

union FragB { unsigned u[8]; v16bf v; };

#if defined(__HIP_DEVICE_COMPILE__) && __has_builtin(__builtin_amdgcn_tensor_load_to_lds)
#define USE_TDM 1
#else
#define USE_TDM 0
#endif

// ---------------- bf16 helpers ----------------------------------------------
__device__ __forceinline__ unsigned short f2bf(float x) {
    unsigned u = __float_as_uint(x);
    unsigned r = u + 0x7FFFu + ((u >> 16) & 1u);   // round-to-nearest-even
    return (unsigned short)(r >> 16);
}
__device__ __forceinline__ float bf2f_bits(unsigned short b) {
    return __uint_as_float(((unsigned)b) << 16);
}

// ---------------- cast fp32 -> bf16 -----------------------------------------
__global__ void castbf_kernel(const float* __restrict__ in,
                              unsigned short* __restrict__ out, int n) {
    int i = blockIdx.x * blockDim.x + threadIdx.x;
    if (i < n) out[i] = f2bf(in[i]);
}

__global__ void zero_kernel(float* p, int n) {
    int i = blockIdx.x * blockDim.x + threadIdx.x;
    if (i < n) p[i] = 0.0f;
}

// ---------------- FPS + sort (one block per batch) --------------------------
__global__ __launch_bounds__(1024)
void fps_kernel(const float* __restrict__ pos, int* __restrict__ cent) {
    __shared__ float dists[NN];
    __shared__ float rv[1024];
    __shared__ int   ri[1024];
    __shared__ int   cl[SS];
    __shared__ int   lastSh;
    const int t = threadIdx.x;
    const int b = blockIdx.x;
    const float* P = pos + (long)b * NN * 3;

    for (int i = t; i < NN; i += 1024) dists[i] = 3.4e38f;
    if (t == 0) cl[0] = 0;
    __syncthreads();

    int last = 0;
    for (int it = 1; it < SS; ++it) {
        float lx = P[last*3+0], ly = P[last*3+1], lz = P[last*3+2];
        float bv = -1.0f; int bi = 0;
        #pragma unroll
        for (int q = 0; q < 4; ++q) {
            int p = t + q * 1024;            // increasing global index order
            float dx = P[p*3+0]-lx, dy = P[p*3+1]-ly, dz = P[p*3+2]-lz;
            float d  = dx*dx + dy*dy + dz*dz;
            float nd = fminf(dists[p], d);
            dists[p] = nd;
            if (nd > bv) { bv = nd; bi = p; } // strict > keeps first index on tie
        }
        rv[t] = bv; ri[t] = bi;
        __syncthreads();
        for (int s = 512; s > 0; s >>= 1) {
            if (t < s) {
                float ov = rv[t+s]; int oi = ri[t+s];
                if (ov > rv[t] || (ov == rv[t] && oi < ri[t])) { rv[t] = ov; ri[t] = oi; }
            }
            __syncthreads();
        }
        if (t == 0) { cl[it] = ri[0]; lastSh = ri[0]; }
        __syncthreads();
        last = lastSh;
    }
    // bitonic ascending sort of cl[0..1023]
    for (int k = 2; k <= SS; k <<= 1) {
        for (int j = k >> 1; j > 0; j >>= 1) {
            int ixj = t ^ j;
            if (ixj > t) {
                int a = cl[t], c = cl[ixj];
                bool up = ((t & k) == 0);
                if ((a > c) == up) { cl[t] = c; cl[ixj] = a; }
            }
            __syncthreads();
        }
    }
    cent[b * SS + t] = cl[t];
}

// ---------------- KNN top-64 (one block per center) -------------------------
__global__ __launch_bounds__(256)
void knn_kernel(const float* __restrict__ pos, const int* __restrict__ cent,
                int* __restrict__ gidx) {
    __shared__ unsigned long long sk[256];
    const int t   = threadIdx.x;
    const int idx = blockIdx.x;            // b*SS + s
    const int b   = idx >> 10;
    const float* P = pos + (long)b * NN * 3;
    int c = cent[idx];
    float cx = P[c*3], cy = P[c*3+1], cz = P[c*3+2];
    float cc = cx*cx + cy*cy + cz*cz;

    unsigned long long keys[16];
    #pragma unroll
    for (int q = 0; q < 16; ++q) {
        int p = t + q * 256;
        float px = P[p*3], py = P[p*3+1], pz = P[p*3+2];
        float dot = cx*px + cy*py + cz*pz;
        float pp  = px*px + py*py + pz*pz;
        float d   = (-2.0f * dot + cc) + pp;     // same expanded form as reference
        unsigned u = __float_as_uint(d);
        unsigned ou = (u & 0x80000000u) ? ~u : (u | 0x80000000u); // total order
        keys[q] = (((unsigned long long)ou) << 32) | (unsigned)p;
    }
    for (int k = 0; k < KK; ++k) {
        unsigned long long lb = ~0ULL;
        #pragma unroll
        for (int q = 0; q < 16; ++q) lb = keys[q] < lb ? keys[q] : lb;
        sk[t] = lb;
        __syncthreads();
        for (int s = 128; s > 0; s >>= 1) {
            if (t < s) { unsigned long long o = sk[t+s]; if (o < sk[t]) sk[t] = o; }
            __syncthreads();
        }
        unsigned long long w = sk[0];
        if (t == 0) gidx[(long)idx * KK + k] = (int)(unsigned)(w & 0xffffffffULL);
        #pragma unroll
        for (int q = 0; q < 16; ++q) if (keys[q] == w) keys[q] = ~0ULL;
        __syncthreads();
    }
}

// ---------------- center_pos gather -----------------------------------------
__global__ void centerpos_kernel(const float* __restrict__ pos,
                                 const int* __restrict__ cent,
                                 float* __restrict__ out) {
    int i = blockIdx.x * blockDim.x + threadIdx.x;
    if (i < BB * SS) {
        int b = i >> 10;
        int c = cent[i];
        const float* p = pos + ((long)(b * NN + c)) * 3;
        out[i*3+0] = p[0]; out[i*3+1] = p[1]; out[i*3+2] = p[2];
    }
}

// ---------------- fused WMMA GEMM (64 rows x 256 cols per block) ------------
// mode 0: A = gathered [src-dst | src] from feat (fp32 -> bf16)
// mode 1: A = relu(h_in*scale + shift) from hbuf bf16 (in-place safe: hout==hin)
__global__ __launch_bounds__(256)
void gemm_kernel(const float* __restrict__ feat,
                 const int* __restrict__ gidx,
                 const int* __restrict__ cent,
                 const unsigned short* __restrict__ hin,
                 const float* __restrict__ scale,
                 const float* __restrict__ shiftv,
                 const unsigned short* __restrict__ Wb,   // bf16 [COUT][CIN]
                 const float* __restrict__ bias,
                 unsigned short* __restrict__ hout,       // bf16 [ROWS][COUT]
                 int mode) {
    __shared__ unsigned short rowc[64 * 258];     // 64 rows x 256 ch (+2 pad)
    __shared__ unsigned short btile[256 * 34];    // 256 ch  x 32 K  (+2 pad)
    const int tid = threadIdx.x;
    const long rowBase = (long)blockIdx.x * 64;

    // prefetch full weight matrix into L2 (global_prefetch_b8); 256 thr x 512B
    __builtin_prefetch(((const char*)Wb) + (size_t)tid * 512, 0, 1);

    // -------- stage A rows into LDS --------
    if (mode == 0) {
        for (int i = tid; i < 64 * 256; i += 256) {
            int m = i >> 8, c = i & 255;
            long r = rowBase + m;
            int b = (int)(r >> 16);                 // S*K = 65536
            int s = (int)((r >> 6) & 1023);
            int src = gidx[r];
            int dst = cent[b * SS + s];
            float v;
            if (c < 128)
                v = feat[((long)(b*NN + src))*FF + c] - feat[((long)(b*NN + dst))*FF + c];
            else
                v = feat[((long)(b*NN + src))*FF + (c - 128)];
            rowc[m * 258 + c] = f2bf(v);
        }
    } else {
        for (int i = tid; i < 64 * 256; i += 256) {
            int m = i >> 8, c = i & 255;
            long r = rowBase + m;
            float v = bf2f_bits(hin[r * 256 + c]) * scale[c] + shiftv[c];
            v = v > 0.0f ? v : 0.0f;
            rowc[m * 258 + c] = f2bf(v);
        }
    }

    const int wave = tid >> 5, lane = tid & 31;
    const int half = lane >> 4, ln16 = lane & 15;
    const int wm = wave & 1;          // row half of block tile (0..1)
    const int wn = wave >> 1;         // col quarter (0..3), 64 cols each

    v8f acc[2][4];
    #pragma unroll
    for (int i2 = 0; i2 < 2; ++i2)
        #pragma unroll
        for (int j4 = 0; j4 < 4; ++j4)
            acc[i2][j4] = (v8f){0.f,0.f,0.f,0.f,0.f,0.f,0.f,0.f};

    const unsigned* rowc_u = (const unsigned*)rowc;
    const unsigned* bt_u   = (const unsigned*)btile;
    const unsigned* Wu     = (const unsigned*)Wb;
    (void)Wu;

    for (int c0 = 0; c0 < CIN; c0 += 32) {
        __syncthreads();
#if USE_TDM
        // -------- TDM: DMA weight k-slice W[0..255][c0..c0+31] into LDS ------
        // pad_enable: +1 DWORD after every 16 DWORDs (64B row) -> LDS row
        // stride 17 dwords = 34 bf16, matching btile[o*34 + c].
        if (wave == 0) {
            unsigned ldsAddr = (unsigned)(unsigned long long)(const void*)btile;
            unsigned long long ga = (unsigned long long)Wb + (unsigned)(c0 * 2);
            u32x4 g0;
            g0.x = 1u;                                        // count=1, user mode
            g0.y = ldsAddr;                                   // lds_addr
            g0.z = (unsigned)(ga & 0xffffffffu);              // global_addr[95:64]
            g0.w = (unsigned)((ga >> 32) & 0x1FFFFFFu)        // global_addr[120:96]
                 | (2u << 30);                                // type = 2 (image)
            i32x8 g1;
            g1[0] = 0x00D10000;   // data_size=2B | pad_enable | pad_interval=16DW
            g1[1] = (int)(256u << 16);   // tensor_dim0 = 256 (low 16 @ [63:48])
            g1[2] = (int)(256u << 16);   // tensor_dim1 = 256 (low 16 @ [95:80])
            g1[3] = (int)(32u  << 16);   // tile_dim0   = 32  (@ [127:112])
            g1[4] = 256;                 // tile_dim1   = 256; tile_dim2 = 0
            g1[5] = 256;                 // tensor_dim0_stride = 256
            g1[6] = 0;                   // stride hi / tensor_dim1_stride = 0
            g1[7] = 0;
            i32x4 gz = (i32x4){0,0,0,0};
#if __clang_major__ >= 23
            i32x8 gz8 = (i32x8){0,0,0,0,0,0,0,0};
            __builtin_amdgcn_tensor_load_to_lds(g0, g1, gz, gz, gz8, 0);
#else
            __builtin_amdgcn_tensor_load_to_lds(g0, g1, gz, gz, 0);
#endif
            __builtin_amdgcn_s_wait_tensorcnt(0);
        }
        __syncthreads();
#else
        // -------- fallback: VALU staging of the weight k-slice --------------
        for (int i = tid; i < 4096; i += 256) {
            int o = i >> 4, cw = i & 15;
            ((unsigned*)btile)[o * 17 + cw] = Wu[o * 128 + (c0 >> 1) + cw];
        }
        __syncthreads();
#endif

        // B fragments: lane ln16 -> col, half -> K 16-block (u[j] = K 2j..2j+1)
        FragB bfr[4];
        #pragma unroll
        for (int j4 = 0; j4 < 4; ++j4) {
            int o = wn * 64 + j4 * 16 + ln16;
            const unsigned* p = bt_u + o * 17;
            #pragma unroll
            for (int j = 0; j < 8; ++j) bfr[j4].u[j] = p[j + half * 8];
        }
        // A fragments: contiguous K layout (half -> K 16-block, matching B)
        FragB af[2];
        #pragma unroll
        for (int i2 = 0; i2 < 2; ++i2) {
            int row = wm * 32 + i2 * 16 + ln16;
            const unsigned* p = rowc_u + row * 129 + (c0 >> 1);
            #pragma unroll
            for (int j = 0; j < 8; ++j)
                af[i2].u[j] = p[j + half * 8];
        }
        #pragma unroll
        for (int i2 = 0; i2 < 2; ++i2)
            #pragma unroll
            for (int j4 = 0; j4 < 4; ++j4)
                acc[i2][j4] = __builtin_amdgcn_wmma_f32_16x16x32_bf16(
                    false, af[i2].v, false, bfr[j4].v,
                    (short)0, acc[i2][j4], false, false);
    }

    // -------- bias + store (C layout: M = p + 8*half, N = ln16) --------
    #pragma unroll
    for (int i2 = 0; i2 < 2; ++i2) {
        #pragma unroll
        for (int j4 = 0; j4 < 4; ++j4) {
            int ch = wn * 64 + j4 * 16 + ln16;
            float bch = bias[ch];
            #pragma unroll
            for (int p = 0; p < 8; ++p) {
                long row = rowBase + wm * 32 + i2 * 16 + p + 8 * half;
                hout[row * 256 + ch] = f2bf(acc[i2][j4][p] + bch);
            }
        }
    }
}

// ---------------- per-channel sum / sumsq over hbuf -------------------------
__global__ __launch_bounds__(128)
void stats_kernel(const unsigned short* __restrict__ h,
                  float* __restrict__ sum, float* __restrict__ sumsq) {
    const int t = threadIdx.x;                        // 128 threads, 2 ch each
    long r0 = (long)blockIdx.x * 2048;
    const unsigned* hu = (const unsigned*)h;
    float s0 = 0.f, s1 = 0.f, q0 = 0.f, q1 = 0.f;
    for (long r = r0; r < r0 + 2048; ++r) {
        unsigned u = hu[r * 128 + t];
        float f0 = __uint_as_float(u << 16);
        float f1 = __uint_as_float(u & 0xFFFF0000u);
        s0 += f0; s1 += f1; q0 += f0 * f0; q1 += f1 * f1;
    }
    atomicAdd(&sum[2*t],   s0); atomicAdd(&sum[2*t+1],   s1);
    atomicAdd(&sumsq[2*t], q0); atomicAdd(&sumsq[2*t+1], q1);
}

__global__ void finalize_kernel(const float* __restrict__ sum,
                                const float* __restrict__ sumsq,
                                const float* __restrict__ gamma,
                                const float* __restrict__ beta,
                                float* __restrict__ scale,
                                float* __restrict__ shiftv) {
    int c = threadIdx.x;
    if (c < COUT) {
        const float inv_n = 1.0f / (float)ROWS;
        float m  = sum[c] * inv_n;
        float v  = sumsq[c] * inv_n - m * m;
        float is = rsqrtf(v + EPSBN);
        float sc = gamma[c] * is;
        scale[c]  = sc;
        shiftv[c] = beta[c] - m * sc;
    }
}

// ---------------- BN2 + ReLU + max over K -----------------------------------
__global__ __launch_bounds__(128)
void maxk_kernel(const unsigned short* __restrict__ h,
                 const float* __restrict__ scale, const float* __restrict__ shiftv,
                 float* __restrict__ out) {
    const int t = threadIdx.x;                       // 128 threads, 2 ch each
    long base = (long)blockIdx.x * KK;
    const unsigned* hu = (const unsigned*)h;
    float sc0 = scale[2*t], sc1 = scale[2*t+1];
    float sh0 = shiftv[2*t], sh1 = shiftv[2*t+1];
    float m0 = 0.f, m1 = 0.f;                        // relu output >= 0
    for (int k = 0; k < KK; ++k) {
        unsigned u = hu[(base + k) * 128 + t];
        float f0 = __uint_as_float(u << 16) * sc0 + sh0;
        float f1 = __uint_as_float(u & 0xFFFF0000u) * sc1 + sh1;
        f0 = f0 > 0.f ? f0 : 0.f;  f1 = f1 > 0.f ? f1 : 0.f;
        m0 = fmaxf(m0, f0);        m1 = fmaxf(m1, f1);
    }
    out[(long)blockIdx.x * 256 + 2*t]     = m0;
    out[(long)blockIdx.x * 256 + 2*t + 1] = m1;
}

// ---------------- host side --------------------------------------------------
extern "C" void kernel_launch(void* const* d_in, const int* in_sizes, int n_in,
                              void* d_out, int out_size, void* d_ws, size_t ws_size,
                              hipStream_t stream) {
    const float* pos    = (const float*)d_in[0];
    const float* feat   = (const float*)d_in[1];
    const float* W1     = (const float*)d_in[2];
    const float* b1     = (const float*)d_in[3];
    const float* gamma1 = (const float*)d_in[4];
    const float* beta1  = (const float*)d_in[5];
    const float* W2     = (const float*)d_in[6];
    const float* b2     = (const float*)d_in[7];
    const float* gamma2 = (const float*)d_in[8];
    const float* beta2  = (const float*)d_in[9];

    // workspace carve-out
    size_t off = 0;
    auto carve = [&](size_t bytes) {
        void* p = (char*)d_ws + off;
        off += (bytes + 255) & ~(size_t)255;
        return p;
    };
    unsigned short* W1b  = (unsigned short*)carve((size_t)COUT*CIN*2);
    unsigned short* W2b  = (unsigned short*)carve((size_t)COUT*COUT*2);
    int*   cent  = (int*)carve((size_t)BB*SS*4);
    int*   gidx  = (int*)carve((size_t)BB*SS*KK*4);
    float* sum   = (float*)carve(512*4);           // sum[256] | sumsq[256]
    float* sumsq = sum + 256;
    float* scale = (float*)carve(256*4);
    float* shiftv= (float*)carve(256*4);
    unsigned short* hbuf = (unsigned short*)carve((size_t)ROWS*COUT*2);

    float* out_pos  = (float*)d_out;               // (B,S,3)
    float* out_feat = out_pos + (size_t)BB*SS*3;   // (B,S,256)

    // 1) weight casts
    castbf_kernel<<<(COUT*CIN+255)/256, 256, 0, stream>>>(W1, W1b, COUT*CIN);
    castbf_kernel<<<(COUT*COUT+255)/256, 256, 0, stream>>>(W2, W2b, COUT*COUT);

    // 2) FPS + sort, KNN, center_pos
    fps_kernel<<<BB, 1024, 0, stream>>>(pos, cent);
    knn_kernel<<<BB*SS, 256, 0, stream>>>(pos, cent, gidx);
    centerpos_kernel<<<(BB*SS+255)/256, 256, 0, stream>>>(pos, cent, out_pos);

    // 3) GEMM1 (gather + WMMA) -> hbuf
    gemm_kernel<<<(int)(ROWS/64), 256, 0, stream>>>(
        feat, gidx, cent, nullptr, nullptr, nullptr, W1b, b1, hbuf, 0);

    // 4) BN1 stats
    zero_kernel<<<1, 512, 0, stream>>>(sum, 512);
    stats_kernel<<<(int)(ROWS/2048), 128, 0, stream>>>(hbuf, sum, sumsq);
    finalize_kernel<<<1, 256, 0, stream>>>(sum, sumsq, gamma1, beta1, scale, shiftv);

    // 5) GEMM2 with fused BN1+ReLU, in-place over hbuf
    gemm_kernel<<<(int)(ROWS/64), 256, 0, stream>>>(
        nullptr, nullptr, nullptr, hbuf, scale, shiftv, W2b, b2, hbuf, 1);

    // 6) BN2 stats
    zero_kernel<<<1, 512, 0, stream>>>(sum, 512);
    stats_kernel<<<(int)(ROWS/2048), 128, 0, stream>>>(hbuf, sum, sumsq);
    finalize_kernel<<<1, 256, 0, stream>>>(sum, sumsq, gamma2, beta2, scale, shiftv);

    // 7) BN2 + ReLU + max over K
    maxk_kernel<<<BB*SS, 128, 0, stream>>>(hbuf, scale, shiftv, out_feat);
}
